// RelationModule_39204461478679
// MI455X (gfx1250) — compile-verified, matrix operand order checked by reading, not run on hardware
//
#include <hip/hip_runtime.h>
#include <hip/hip_bf16.h>

// Problem constants (from reference)
#define C_    2
#define N_    1024
#define D_    1024
#define G_    16
#define DG_   64
#define FC0_  64
#define TOPK_ 10
#define JS_   4            // j-range splits per (c, i-tile) for occupancy

typedef __attribute__((ext_vector_type(16))) __bf16 v16bf;
typedef __attribute__((ext_vector_type(8)))  __bf16 v8bf;
typedef __attribute__((ext_vector_type(8)))  float  v8f;

static __device__ __forceinline__ __bf16 f2bf(float f) {
  unsigned u = __builtin_bit_cast(unsigned, f);
  u += 0x7FFFu + ((u >> 16) & 1u);          // round-to-nearest-even
  unsigned short s = (unsigned short)(u >> 16);
  return __builtin_bit_cast(__bf16, s);
}

// Natural log for inputs known to be >= 1e-6 (no denormal guard needed):
// raw v_log_f32 (log2) * ln(2).
static __device__ __forceinline__ float fast_ln(float x) {
  return __builtin_amdgcn_logf(x) * 0.69314718056f;
}

static __device__ __forceinline__ v8f wmma_bf16(v16bf a, v16bf b, v8f c) {
  return __builtin_amdgcn_wmma_f32_16x16x32_bf16(false, a, false, b, (short)0, c,
                                                 false, false);
}

// Load 16 f32 (runs p[0..7], p[16..23]) as bf16 in WMMA lane layout using
// vector loads + packed fptrunc (v_cvt_pk_bf16_f32 on gfx1250).
// Caller pre-offsets p by the lane's klo (0 or 8); both runs are 32B-aligned.
static __device__ __forceinline__ v16bf load16_f32_as_bf16(const float* __restrict__ p) {
  v8f lo = *reinterpret_cast<const v8f*>(p);
  v8f hi = *reinterpret_cast<const v8f*>(p + 16);
  v8bf blo = __builtin_convertvector(lo, v8bf);
  v8bf bhi = __builtin_convertvector(hi, v8bf);
  v16bf r;
#pragma unroll
  for (int i = 0; i < 8; ++i) { r[i] = blo[i]; r[8 + i] = bhi[i]; }
  return r;
}

static __device__ __forceinline__ v16bf load16_bf16(const __bf16* __restrict__ p) {
  v8bf lo = *reinterpret_cast<const v8bf*>(p);
  v8bf hi = *reinterpret_cast<const v8bf*>(p + 16);
  v16bf r;
#pragma unroll
  for (int i = 0; i < 8; ++i) { r[i] = lo[i]; r[8 + i] = hi[i]; }
  return r;
}

// ---------------------------------------------------------------------------
// K1: Y = A @ W^T (+ bias)   A = fa viewed [C*N, D] (fa[c,n,d] = f_a[n,c,d])
// MODE 0: bf16 out (Q/K, +bias); MODE 1: f32 out (V, no bias)
// grid: 1024 blocks x 256 thr (8 waves); wave -> one 16x16 tile, K=1024.
// ---------------------------------------------------------------------------
template <int MODE>
__global__ void k_gemm(const float* __restrict__ f_a,   // [N, C, D]
                       const float* __restrict__ W,     // [1024,1024] row=out ch
                       const float* __restrict__ bias,  // [1024] (MODE 0)
                       __bf16* __restrict__ out_bf,
                       float*  __restrict__ out_f) {
  const int lane = threadIdx.x & 31;
  const int wave = threadIdx.x >> 5;
  const int wg   = blockIdx.x * 8 + wave;      // 8192 tiles
  const int row0 = (wg >> 6) * 16;
  const int col0 = (wg & 63) * 16;
  const int lm   = lane & 15;
  const int klo  = (lane >> 4) * 8;

  const int arow = row0 + lm;                  // flattened (c*N + n)
  const int c  = arow >> 10;
  const int nn = arow & 1023;
  const float* pa = f_a + ((size_t)nn * C_ + c) * D_ + klo;
  const float* pb = W   + (size_t)(col0 + lm) * D_ + klo;

  v8f acc = {};
  for (int k0 = 0; k0 < D_; k0 += 32)
    acc = wmma_bf16(load16_f32_as_bf16(pa + k0), load16_f32_as_bf16(pb + k0), acc);

  const int ccol = col0 + lm;
  const float bv = (MODE == 0) ? bias[ccol] : 0.0f;
  const int mhi = (lane >> 4) * 8;
#pragma unroll
  for (int r = 0; r < 8; ++r) {
    const int m = row0 + mhi + r;
    const float v = acc[r] + bv;
    if (MODE == 0) out_bf[(size_t)m * 1024 + ccol] = f2bf(v);
    else           out_f [(size_t)m * 1024 + ccol] = v;
  }
}

// ---------------------------------------------------------------------------
// K2 (fused): per block = (c, i-tile of 16, j-split of 256 columns).
// Streams 16 j-tiles; per tile each of 8 waves does:
//   - geo-bias WMMAs for 2 i-rows (rows=j, cols=16 groups, K=64) -> log -> LDS
//   - affinity WMMAs for 2 groups (Q_i x K_j^T, K=64) -> /8 -> LDS
// then 256 threads (one per (g,i) row) fold 16 logits into a sorted top-10.
// Prefetches next tile's PE/K lines to overlap the combine phase.
// Emits per-split top-10 (val, idx) to workspace.
// ---------------------------------------------------------------------------
__global__ void k_fused(const float* __restrict__ pe,    // [C,N,N,64]
                        const float* __restrict__ wg_w,  // [16,64]
                        const float* __restrict__ wg_b,  // [16]
                        const float* __restrict__ iou,   // [C,N,N]
                        const __bf16* __restrict__ Q,    // [C*N,1024]
                        const __bf16* __restrict__ Kb,   // [C*N,1024]
                        float* __restrict__ pval,        // [32768, JS_, 10]
                        int*   __restrict__ pidx) {
  __shared__ float gbL [16][16][17];   // [i_local][j_local][g]
  __shared__ float affL[16][16][17];   // [g][i_local][j_local]
  __shared__ float iouL[16][16];       // [i_local][j_local]

  const int tid  = threadIdx.x;
  const int lane = tid & 31;
  const int wave = tid >> 5;
  const int lm   = lane & 15;
  const int klo  = (lane >> 4) * 8;
  const int mhi  = (lane >> 4) * 8;

  const int blk = blockIdx.x;                  // C * 64 * JS_ = 512
  const int c   = blk / (64 * JS_);
  const int rm  = blk % (64 * JS_);
  const int it  = rm / JS_;                    // i-tile 0..63
  const int js  = rm % JS_;                    // j-split 0..3
  const int i0  = it * 16;
  const int jbase = js * (N_ / JS_);           // 256-wide j range
  const int NT = (N_ / JS_) / 16;              // 16 j-tiles

  // Preload Q A-fragments for this wave's two groups (rows = i).
  const int g0 = wave * 2;
  v16bf aQ[2][2];
#pragma unroll
  for (int gg = 0; gg < 2; ++gg) {
    const __bf16* pq = Q + ((size_t)(c * N_ + i0 + lm)) * 1024 + (g0 + gg) * 64 + klo;
    aQ[gg][0] = load16_bf16(pq);
    aQ[gg][1] = load16_bf16(pq + 32);
  }
  // Preload WG_w B-fragments (cols = g).
  const float* pwg = wg_w + lm * 64 + klo;
  const v16bf bG0 = load16_f32_as_bf16(pwg);
  const v16bf bG1 = load16_f32_as_bf16(pwg + 32);
  const float bvG = wg_b[lm];

  // Running sorted top-10 for this thread's row (g = tid>>4, i_local = tid&15).
  const int myg = tid >> 4;
  const int myi = tid & 15;
  float tv[TOPK_]; int tj[TOPK_];
#pragma unroll
  for (int t = 0; t < TOPK_; ++t) { tv[t] = -3.0e38f; tj[t] = 0x7FFFFFFF; }

  const float LOG1EM6 = -13.815510558f;

  for (int jt = 0; jt < NT; ++jt) {
    const int j0 = jbase + jt * 16;

    // --- geo bias: rows = j, cols = g, for i rows {2*wave, 2*wave+1} ---
#pragma unroll
    for (int rr = 0; rr < 2; ++rr) {
      const int il = wave * 2 + rr;
      const float* pa = pe + (((size_t)(c * N_ + (i0 + il)) * N_) + (j0 + lm)) * 64 + klo;
      if (jt + 1 < NT) {                       // prefetch next tile's PE lines
        __builtin_prefetch(pa + 1024, 0, 3);   // j advances by 16 rows of 64 f32
        __builtin_prefetch(pa + 1024 + 32, 0, 3);
      }
      v8f acc = {};
      acc = wmma_bf16(load16_f32_as_bf16(pa),      bG0, acc);
      acc = wmma_bf16(load16_f32_as_bf16(pa + 32), bG1, acc);
#pragma unroll
      for (int r = 0; r < 8; ++r) {
        const int jl = mhi + r;
        float v = acc[r] + bvG;                 // bias; lane col = group lm
        v = v > 1e-6f ? v : 1e-6f;              // relu + clip
        gbL[il][jl][lm] = fast_ln(v);           // v_log_f32 * ln2 (no denorm guard)
      }
    }
    // --- affinity: Q_i x K_j^T for groups {2*wave, 2*wave+1} ---
#pragma unroll
    for (int gg = 0; gg < 2; ++gg) {
      const __bf16* pk = Kb + ((size_t)(c * N_ + j0 + lm)) * 1024 + (g0 + gg) * 64 + klo;
      if (jt + 1 < NT)
        __builtin_prefetch(pk + 16 * 1024, 0, 3);
      v8f acc = {};
      acc = wmma_bf16(aQ[gg][0], load16_bf16(pk),      acc);
      acc = wmma_bf16(aQ[gg][1], load16_bf16(pk + 32), acc);
#pragma unroll
      for (int r = 0; r < 8; ++r)
        affL[g0 + gg][mhi + r][lm] = acc[r] * 0.125f;   // lane col = j
    }
    // --- iou term: one element per thread ---
    {
      const int il = tid >> 4, jl = tid & 15;
      const float u = iou[((size_t)c * N_ + (i0 + il)) * N_ + (j0 + jl)];
      iouL[il][jl] = (u >= 1e-6f) ? 0.0f : LOG1EM6;
    }
    __syncthreads();

    // --- combine + running top-10 (thread owns row (myg, myi)) ---
#pragma unroll
    for (int jl = 0; jl < 16; ++jl) {
      const float w = gbL[myi][jl][myg] + affL[myg][myi][jl] + iouL[myi][jl];
      if (w > tv[TOPK_ - 1]) {                 // strict > : first-occurrence ties
        float v = w; int j = j0 + jl;
#pragma unroll
        for (int p = 0; p < TOPK_; ++p) {
          if (v > tv[p]) { float t0 = tv[p]; int t1 = tj[p];
                           tv[p] = v; tj[p] = j; v = t0; j = t1; }
        }
      }
    }
    __syncthreads();
  }

  // Emit this split's top-10.
  const int row = (c * G_ + myg) * N_ + (i0 + myi);     // 0..32767
  const size_t base = ((size_t)row * JS_ + js) * TOPK_;
#pragma unroll
  for (int t = 0; t < TOPK_; ++t) { pval[base + t] = tv[t]; pidx[base + t] = tj[t]; }
}

// ---------------------------------------------------------------------------
// K3: merge JS_ sorted top-10 lists per (c,g,i) row, softmax, gather V rows,
// add conv bias, store. One wave per row; grid 4096 x 256.
// ---------------------------------------------------------------------------
__global__ void k_merge(const float* __restrict__ pval,   // [32768, JS_, 10]
                        const int*   __restrict__ pidx,
                        const float* __restrict__ V,      // [C*N, 1024]
                        const float* __restrict__ conv_b, // [1024]
                        float* __restrict__ out) {        // [N, C, 1024]
  const int lane = threadIdx.x & 31;
  const int wave = threadIdx.x >> 5;
  const int row = blockIdx.x * 8 + wave;        // (c*16+g)*1024+i
  const int c = row >> 14;
  const int g = (row >> 10) & 15;
  const int i = row & 1023;

  float tv[TOPK_]; int tj[TOPK_];
#pragma unroll
  for (int t = 0; t < TOPK_; ++t) { tv[t] = -3.0e38f; tj[t] = 0x7FFFFFFF; }

  // Insert all candidates; splits visited in ascending-j order, so strict >
  // insertion preserves jax.top_k first-occurrence tie ordering.
#pragma unroll
  for (int s = 0; s < JS_; ++s) {
    const size_t base = ((size_t)row * JS_ + s) * TOPK_;
#pragma unroll
    for (int t = 0; t < TOPK_; ++t) {
      float v = pval[base + t]; int j = pidx[base + t];
      if (v > tv[TOPK_ - 1]) {
#pragma unroll
        for (int p = 0; p < TOPK_; ++p) {
          if (v > tv[p]) { float t0 = tv[p]; int t1 = tj[p];
                           tv[p] = v; tj[p] = j; v = t0; j = t1; }
        }
      }
    }
  }

  const float mx = tv[0];
  float sf[TOPK_]; float s = 0.0f;
#pragma unroll
  for (int t = 0; t < TOPK_; ++t) { sf[t] = __expf(tv[t] - mx); s += sf[t]; }
  const float inv = 1.0f / s;

  const int col0 = g * 64;
#pragma unroll
  for (int oo = 0; oo < 2; ++oo) {
    const int o = lane + 32 * oo;
    float acc = conv_b[col0 + o];
#pragma unroll
    for (int t = 0; t < TOPK_; ++t)
      acc += sf[t] * inv * V[((size_t)(c * N_ + tj[t])) * 1024 + col0 + o];
    out[(((size_t)i) * C_ + c) * 1024 + col0 + o] = acc;
  }
}

// ---------------------------------------------------------------------------
extern "C" void kernel_launch(void* const* d_in, const int* in_sizes, int n_in,
                              void* d_out, int out_size, void* d_ws, size_t ws_size,
                              hipStream_t stream) {
  (void)in_sizes; (void)n_in; (void)out_size; (void)ws_size;
  const float* f_a    = (const float*)d_in[0];
  const float* pe     = (const float*)d_in[1];
  const float* iou    = (const float*)d_in[2];
  const float* WG_w   = (const float*)d_in[3];
  const float* WG_b   = (const float*)d_in[4];
  const float* WK_w   = (const float*)d_in[5];
  const float* WK_b   = (const float*)d_in[6];
  const float* WQ_w   = (const float*)d_in[7];
  const float* WQ_b   = (const float*)d_in[8];
  const float* conv_w = (const float*)d_in[9];
  const float* conv_b = (const float*)d_in[10];

  char* ws = (char*)d_ws;
  __bf16* Q    = (__bf16*)ws;                         //  4 MB
  __bf16* Kb   = (__bf16*)(ws + (4u  << 20));         //  4 MB
  float*  V    = (float*) (ws + (8u  << 20));         //  8 MB
  float*  pval = (float*) (ws + (16u << 20));         //  5.25 MB
  int*    pidx = (int*)   (ws + (22u << 20));         //  5.25 MB
  float*  out  = (float*)d_out;

  const dim3 blk(256);
  k_gemm<0><<<1024, blk, 0, stream>>>(f_a, WQ_w, WQ_b, Q,  nullptr);
  k_gemm<0><<<1024, blk, 0, stream>>>(f_a, WK_w, WK_b, Kb, nullptr);
  k_gemm<1><<<1024, blk, 0, stream>>>(f_a, conv_w, nullptr, nullptr, V);
  k_fused<<<C_ * 64 * JS_, blk, 0, stream>>>(pe, WG_w, WG_b, iou, Q, Kb, pval, pidx);
  k_merge<<<4096, blk, 0, stream>>>(pval, pidx, V, conv_b, out);
}
